// MultiHeadAttension_63084479643820
// MI455X (gfx1250) — matrix operand order repaired and checked by reading, not compiled, not run on hardware
//
#include <hip/hip_runtime.h>
#include <hip/hip_bf16.h>
#include <stdint.h>

// MI455X / gfx1250 multi-head causal attention, bf16 WMMA path.
// B=4, S=2048, D=1024, H=16, HD=64.
//
// K1: fused QKV GEMM (128x64x64 tiles, 8 WMMA/wave/iter), Q pre-scaled 1/8.
// K2: flash attention, 128-row q tiles, in-register softmax reductions,
//     GLOBAL_LOAD_ASYNC_TO_LDS staging for Q/K tiles.
// K3: output projection + bias.

typedef __attribute__((ext_vector_type(16))) __bf16 v16bf;
typedef __attribute__((ext_vector_type(8)))  __bf16 v8bf;
typedef __attribute__((ext_vector_type(8)))  float  v8f;

#define B_  4
#define S_  2048
#define D_  1024
#define H_  16
#define HD_ 64

#define WMMA_BF16(a, b, c) \
  __builtin_amdgcn_wmma_f32_16x16x32_bf16(false, (a), false, (b), (short)0, (c), false, false)

// ---- gfx1250 async global->LDS staging (ASYNCcnt path) --------------------
// Toolchain-probed signature: (v4i AS1*, v4i AS3*, imm offset, imm cpol).
#if defined(__gfx1250__) && \
    __has_builtin(__builtin_amdgcn_global_load_async_to_lds_b128) && \
    __has_builtin(__builtin_amdgcn_s_wait_asynccnt)
#define ASYNC_LDS 1
typedef int v4i __attribute__((vector_size(16)));
typedef __attribute__((address_space(1))) v4i as1_v4i;
typedef __attribute__((address_space(3))) v4i as3_v4i;
__device__ __forceinline__ void async_cp16(void* lds, const void* g) {
  __builtin_amdgcn_global_load_async_to_lds_b128(
      (as1_v4i*)(uintptr_t)g,
      (as3_v4i*)(unsigned)(uintptr_t)lds, 0, 0);
}
__device__ __forceinline__ void async_wait0() {
  __builtin_amdgcn_s_wait_asynccnt(0);
}
#else
#define ASYNC_LDS 0
#endif

union V16U { v16bf v; v8bf h[2]; };

// A-matrix 16x32 bf16 fragment (ISA 7.12.2): elements 0..7 -> K = 8*lh + e,
// elements 8..15 -> K = 8*lh + 16 + (e-8).
__device__ __forceinline__ v16bf ldsA_frag(const __bf16* rowPtr, int kbase, int lh) {
  V16U u;
  u.h[0] = *(const v8bf*)(rowPtr + kbase + 8 * lh);
  u.h[1] = *(const v8bf*)(rowPtr + kbase + 8 * lh + 16);
  return u.v;
}

// B-matrix 32x16 bf16 fragment: N = lane&15 (row pointer), K order matches A:
// elements 0..7 -> K = 8*lh + e, elements 8..15 -> K = 8*lh + 16 + (e-8).
__device__ __forceinline__ v16bf ldsB_frag(const __bf16* rowPtr, int kbase, int lh) {
  V16U u;
  u.h[0] = *(const v8bf*)(rowPtr + kbase + 8 * lh);
  u.h[1] = *(const v8bf*)(rowPtr + kbase + 8 * lh + 16);
  return u.v;
}

__device__ __forceinline__ v8bf cvt8(float4 f0, float4 f1) {
  v8bf v;
  v[0] = (__bf16)f0.x; v[1] = (__bf16)f0.y; v[2] = (__bf16)f0.z; v[3] = (__bf16)f0.w;
  v[4] = (__bf16)f1.x; v[5] = (__bf16)f1.y; v[6] = (__bf16)f1.z; v[7] = (__bf16)f1.w;
  return v;
}

// ---------------------------------------------------------------------------
// Kernel 1: fused QKV projection GEMM. out = x @ W -> bf16 [B,H,S,HD].
// Block tile 128x64, BK=64; wave tile 32x32 (4 acc, 8 WMMA per K block).
// ---------------------------------------------------------------------------
__global__ __launch_bounds__(256) void qkv_gemm(
    const float* __restrict__ x,
    const float* __restrict__ Wq, const float* __restrict__ Wk,
    const float* __restrict__ Wv,
    __bf16* __restrict__ Qo, __bf16* __restrict__ Ko, __bf16* __restrict__ Vo) {
  __shared__ __align__(32) __bf16 As[128][64];  // [row][k]
  __shared__ __align__(32) __bf16 Bs[64][64];   // [n][k] (transposed on fill)
  const int t    = threadIdx.x;
  const int lane = t & 31, wave = t >> 5;
  const int lh   = lane >> 4, lm = lane & 15;
  const int wm   = wave & 3,  wn = wave >> 2;
  const int rowBase = blockIdx.y * 128;
  const int colBase = blockIdx.x * 64;
  const float* W = (blockIdx.z == 0) ? Wq : (blockIdx.z == 1) ? Wk : Wv;

  v8f acc[2][2] = {{{}, {}}, {{}, {}}};
  for (int kk = 0; kk < D_; kk += 64) {
#pragma unroll
    for (int g = 0; g < 4; ++g) {               // 128x64 x-tile: b128 + cvt_pk
      int idx = t + 256 * g;                    // 0..1023, 8 elems each
      int r = idx >> 3, c = (idx & 7) * 8;
      const float4* src = (const float4*)&x[(size_t)(rowBase + r) * D_ + kk + c];
      *(v8bf*)&As[r][c] = cvt8(src[0], src[1]);
    }
#pragma unroll
    for (int g = 0; g < 2; ++g) {               // 64x64 W-tile, transposed fill
      int idx = t + 256 * g;                    // 0..511
      int k = idx >> 3, n0 = (idx & 7) * 8;
      const float4* src = (const float4*)&W[(size_t)(kk + k) * D_ + colBase + n0];
      v8bf e = cvt8(src[0], src[1]);
#pragma unroll
      for (int j = 0; j < 8; ++j) Bs[n0 + j][k] = e[j];
    }
    __syncthreads();
#pragma unroll
    for (int ks = 0; ks < 64; ks += 32) {
      v16bf a0 = ldsA_frag(&As[wm * 32 + lm][0],      ks, lh);
      v16bf a1 = ldsA_frag(&As[wm * 32 + 16 + lm][0], ks, lh);
      v16bf b0 = ldsB_frag(&Bs[wn * 32 + lm][0],      ks, lh);
      v16bf b1 = ldsB_frag(&Bs[wn * 32 + 16 + lm][0], ks, lh);
      acc[0][0] = WMMA_BF16(a0, b0, acc[0][0]);
      acc[0][1] = WMMA_BF16(a0, b1, acc[0][1]);
      acc[1][0] = WMMA_BF16(a1, b0, acc[1][0]);
      acc[1][1] = WMMA_BF16(a1, b1, acc[1][1]);
    }
    __syncthreads();
  }

  const float scale = (blockIdx.z == 0) ? 0.125f : 1.0f;  // 1/sqrt(64) on Q
  __bf16* dst = (blockIdx.z == 0) ? Qo : (blockIdx.z == 1) ? Ko : Vo;
  const int h = colBase >> 6;                   // one head per 64-col tile
#pragma unroll
  for (int mi = 0; mi < 2; ++mi) {
#pragma unroll
    for (int r = 0; r < 8; ++r) {
      int row = rowBase + wm * 32 + mi * 16 + r + 8 * lh;  // M = vgpr + 8*lh
      int b = row >> 11, s = row & (S_ - 1);
      size_t o = (((size_t)(b * H_ + h)) * S_ + s) * HD_;
      int hd0 = wn * 32 + lm, hd1 = hd0 + 16;
      dst[o + hd0] = (__bf16)(acc[mi][0][r] * scale);
      dst[o + hd1] = (__bf16)(acc[mi][1][r] * scale);
    }
  }
}

// ---------------------------------------------------------------------------
// Kernel 2: flash-style causal attention, 128-row q tile per block.
// scores = Q K^T (Q pre-scaled); in-register online softmax; ctx += P V.
// ---------------------------------------------------------------------------
__global__ __launch_bounds__(256) void attn(
    const __bf16* __restrict__ Q, const __bf16* __restrict__ K,
    const __bf16* __restrict__ V, __bf16* __restrict__ CTX) {
  __shared__ __align__(32) __bf16 Qs[128][64];  // [q][hd]
  __shared__ __align__(32) __bf16 Ks[64][64];   // [key][hd]  (B-side of QK^T)
  __shared__ __align__(32) __bf16 Vt[64][64];   // [hd][key]  (B-side of PV)
  __shared__ __align__(32) __bf16 Ps[128][64];  // probs      (A-side of PV)
  __shared__ float mrow[128], lrow[128], frow[128];
  __shared__ float wmax[2][128], wsum[2][128];  // per-wn partials

  const int t    = threadIdx.x;
  const int lane = t & 31, wave = t >> 5;
  const int lh   = lane >> 4, lm = lane & 15;
  const int wm   = wave & 3,  wn = wave >> 2;
  const int qBase = blockIdx.x * 128;
  const int bh    = blockIdx.y;
  const size_t base = (size_t)bh * S_ * HD_;

#pragma unroll
  for (int g = 0; g < 4; ++g) {                 // stage 128x64 Q tile
    int idx = t + 256 * g;
    int r = idx >> 3, c = (idx & 7) * 8;
    const __bf16* src = &Q[base + (size_t)(qBase + r) * HD_ + c];
#if ASYNC_LDS
    async_cp16(&Qs[r][c], src);
#else
    *(v8bf*)&Qs[r][c] = *(const v8bf*)src;
#endif
  }
  if (t < 128) { mrow[t] = -3.0e38f; lrow[t] = 0.0f; }
  v8f cacc[2][2] = {{{}, {}}, {{}, {}}};
#if ASYNC_LDS
  async_wait0();
#endif
  __syncthreads();

  const int nkt = 2 * blockIdx.x + 2;           // causal: tiles covering <= qBase+127
  for (int jt = 0; jt < nkt; ++jt) {
#pragma unroll
    for (int g = 0; g < 2; ++g) {               // stage K row-major, V transposed
      int idx = t + 256 * g;
      int r = idx >> 3, c = (idx & 7) * 8;
      const __bf16* ksrc = &K[base + (size_t)(jt * 64 + r) * HD_ + c];
#if ASYNC_LDS
      async_cp16(&Ks[r][c], ksrc);
#else
      *(v8bf*)&Ks[r][c] = *(const v8bf*)ksrc;
#endif
      v8bf vv = *(const v8bf*)&V[base + (size_t)(jt * 64 + r) * HD_ + c];
#pragma unroll
      for (int j = 0; j < 8; ++j) Vt[c + j][r] = vv[j];
    }
#if ASYNC_LDS
    async_wait0();
#endif
    __syncthreads();

    v8f s[2][2] = {{{}, {}}, {{}, {}}};
#pragma unroll
    for (int ks = 0; ks < 64; ks += 32) {       // K-dim = HD = 64
      v16bf a0 = ldsA_frag(&Qs[wm * 32 + lm][0],      ks, lh);
      v16bf a1 = ldsA_frag(&Qs[wm * 32 + 16 + lm][0], ks, lh);
      v16bf b0 = ldsB_frag(&Ks[wn * 32 + lm][0],      ks, lh);
      v16bf b1 = ldsB_frag(&Ks[wn * 32 + 16 + lm][0], ks, lh);
      s[0][0] = WMMA_BF16(a0, b0, s[0][0]);
      s[0][1] = WMMA_BF16(a0, b1, s[0][1]);
      s[1][0] = WMMA_BF16(a1, b0, s[1][0]);
      s[1][1] = WMMA_BF16(a1, b1, s[1][1]);
    }

    // causal mask + wave-local row max (rows live in one 16-lane half)
#pragma unroll
    for (int mi = 0; mi < 2; ++mi) {
#pragma unroll
      for (int r = 0; r < 8; ++r) {
        int row = wm * 32 + mi * 16 + r + 8 * lh;
        int gq  = qBase + row;
        int col0 = wn * 32 + lm, col1 = col0 + 16;
        if (jt * 64 + col0 > gq) s[mi][0][r] = -3.0e38f;
        if (jt * 64 + col1 > gq) s[mi][1][r] = -3.0e38f;
        float v = fmaxf(s[mi][0][r], s[mi][1][r]);
#pragma unroll
        for (int d = 1; d < 16; d <<= 1) v = fmaxf(v, __shfl_xor(v, d, 32));
        if (lm == 0) wmax[wn][row] = v;
      }
    }
    __syncthreads();
    if (t < 128) {                              // combine partials, new max
      float mo = mrow[t];
      float mn = fmaxf(mo, fmaxf(wmax[0][t], wmax[1][t]));
      mrow[t] = mn;
      frow[t] = __expf(mo - mn);
    }
    __syncthreads();

    // p = exp(s - m_new): write probs (bf16) + per-wave row-sum partial
#pragma unroll
    for (int mi = 0; mi < 2; ++mi) {
#pragma unroll
      for (int r = 0; r < 8; ++r) {
        int row = wm * 32 + mi * 16 + r + 8 * lh;
        float mn = mrow[row];
        float p0 = __expf(s[mi][0][r] - mn);
        float p1 = __expf(s[mi][1][r] - mn);
        int col0 = wn * 32 + lm, col1 = col0 + 16;
        Ps[row][col0] = (__bf16)p0;
        Ps[row][col1] = (__bf16)p1;
        float v = p0 + p1;
#pragma unroll
        for (int d = 1; d < 16; d <<= 1) v += __shfl_xor(v, d, 32);
        if (lm == 0) wsum[wn][row] = v;
      }
    }
    __syncthreads();
    if (t < 128) lrow[t] = lrow[t] * frow[t] + wsum[0][t] + wsum[1][t];

    // rescale running ctx and accumulate P @ V
#pragma unroll
    for (int mi = 0; mi < 2; ++mi) {
#pragma unroll
      for (int r = 0; r < 8; ++r) {
        float f = frow[wm * 32 + mi * 16 + r + 8 * lh];
        cacc[mi][0][r] *= f;
        cacc[mi][1][r] *= f;
      }
    }
#pragma unroll
    for (int ks = 0; ks < 64; ks += 32) {       // K-dim = 64 keys
      v16bf a0 = ldsA_frag(&Ps[wm * 32 + lm][0],      ks, lh);
      v16bf a1 = ldsA_frag(&Ps[wm * 32 + 16 + lm][0], ks, lh);
      v16bf b0 = ldsB_frag(&Vt[wn * 32 + lm][0],      ks, lh);
      v16bf b1 = ldsB_frag(&Vt[wn * 32 + 16 + lm][0], ks, lh);
      cacc[0][0] = WMMA_BF16(a0, b0, cacc[0][0]);
      cacc[0][1] = WMMA_BF16(a0, b1, cacc[0][1]);
      cacc[1][0] = WMMA_BF16(a1, b0, cacc[1][0]);
      cacc[1][1] = WMMA_BF16(a1, b1, cacc[1][1]);
    }
    __syncthreads();                            // protect Ks/Vt/Ps for next tile
  }

  const int b = bh >> 4, h = bh & 15;
#pragma unroll
  for (int mi = 0; mi < 2; ++mi) {
#pragma unroll
    for (int r = 0; r < 8; ++r) {
      int row = wm * 32 + mi * 16 + r + 8 * lh;
      float inv = 1.0f / lrow[row];
      int sq = qBase + row;
      int hd0 = wn * 32 + lm, hd1 = hd0 + 16;
      size_t o = ((size_t)(b * S_) + sq) * D_ + h * HD_;
      CTX[o + hd0] = (__bf16)(cacc[mi][0][r] * inv);
      CTX[o + hd1] = (__bf16)(cacc[mi][1][r] * inv);
    }
  }
}

// ---------------------------------------------------------------------------
// Kernel 3: output projection. out = ctx @ Wo + bo (fp32 out).
// Same 128x64x64 tiling; A-tile is bf16 (async-staged).
// ---------------------------------------------------------------------------
__global__ __launch_bounds__(256) void out_proj(
    const __bf16* __restrict__ CTX, const float* __restrict__ Wo,
    const float* __restrict__ bo, float* __restrict__ out) {
  __shared__ __align__(32) __bf16 As[128][64];
  __shared__ __align__(32) __bf16 Bs[64][64];
  const int t    = threadIdx.x;
  const int lane = t & 31, wave = t >> 5;
  const int lh   = lane >> 4, lm = lane & 15;
  const int wm   = wave & 3,  wn = wave >> 2;
  const int rowBase = blockIdx.y * 128;
  const int colBase = blockIdx.x * 64;

  v8f acc[2][2] = {{{}, {}}, {{}, {}}};
  for (int kk = 0; kk < D_; kk += 64) {
#pragma unroll
    for (int g = 0; g < 4; ++g) {               // 128x64 ctx tile (bf16)
      int idx = t + 256 * g;
      int r = idx >> 3, c = (idx & 7) * 8;
      const __bf16* src = &CTX[(size_t)(rowBase + r) * D_ + kk + c];
#if ASYNC_LDS
      async_cp16(&As[r][c], src);
#else
      *(v8bf*)&As[r][c] = *(const v8bf*)src;
#endif
    }
#pragma unroll
    for (int g = 0; g < 2; ++g) {               // 64x64 Wo tile, transposed
      int idx = t + 256 * g;
      int k = idx >> 3, n0 = (idx & 7) * 8;
      const float4* src = (const float4*)&Wo[(size_t)(kk + k) * D_ + colBase + n0];
      v8bf e = cvt8(src[0], src[1]);
#pragma unroll
      for (int j = 0; j < 8; ++j) Bs[n0 + j][k] = e[j];
    }
#if ASYNC_LDS
    async_wait0();
#endif
    __syncthreads();
#pragma unroll
    for (int ks = 0; ks < 64; ks += 32) {
      v16bf a0 = ldsA_frag(&As[wm * 32 + lm][0],      ks, lh);
      v16bf a1 = ldsA_frag(&As[wm * 32 + 16 + lm][0], ks, lh);
      v16bf b0 = ldsB_frag(&Bs[wn * 32 + lm][0],      ks, lh);
      v16bf b1 = ldsB_frag(&Bs[wn * 32 + 16 + lm][0], ks, lh);
      acc[0][0] = WMMA_BF16(a0, b0, acc[0][0]);
      acc[0][1] = WMMA_BF16(a0, b1, acc[0][1]);
      acc[1][0] = WMMA_BF16(a1, b0, acc[1][0]);
      acc[1][1] = WMMA_BF16(a1, b1, acc[1][1]);
    }
    __syncthreads();
  }

#pragma unroll
  for (int mi = 0; mi < 2; ++mi) {
#pragma unroll
    for (int r = 0; r < 8; ++r) {
      int row  = rowBase + wm * 32 + mi * 16 + r + 8 * lh;
      int col0 = colBase + wn * 32 + lm;
      int col1 = col0 + 16;
      out[(size_t)row * D_ + col0] = acc[mi][0][r] + bo[col0];
      out[(size_t)row * D_ + col1] = acc[mi][1][r] + bo[col1];
    }
  }
}

// ---------------------------------------------------------------------------
extern "C" void kernel_launch(void* const* d_in, const int* in_sizes, int n_in,
                              void* d_out, int out_size, void* d_ws, size_t ws_size,
                              hipStream_t stream) {
  (void)in_sizes; (void)n_in; (void)out_size; (void)ws_size;
  const float* x  = (const float*)d_in[0];
  const float* Wq = (const float*)d_in[1];
  const float* Wk = (const float*)d_in[2];
  const float* Wv = (const float*)d_in[3];
  const float* Wo = (const float*)d_in[4];
  const float* bo = (const float*)d_in[5];
  float* out = (float*)d_out;

  // Workspace: Q,K,V head-major bf16 + ctx bf16. 4 * 16 MB = 64 MB.
  const size_t n = (size_t)B_ * H_ * S_ * HD_;  // 8,388,608 elements
  __bf16* Q   = (__bf16*)d_ws;
  __bf16* K   = Q + n;
  __bf16* V   = K + n;
  __bf16* CTX = V + n;

  dim3 g1(D_ / 64, (B_ * S_) / 128, 3);         // 16 x 64 x 3
  qkv_gemm<<<g1, 256, 0, stream>>>(x, Wq, Wk, Wv, Q, K, V);

  dim3 g2(S_ / 128, B_ * H_);                   // 16 x 64
  attn<<<g2, 256, 0, stream>>>(Q, K, V, CTX);

  dim3 g3(D_ / 64, (B_ * S_) / 128);            // 16 x 64
  out_proj<<<g3, 256, 0, stream>>>(CTX, Wo, bo, out);
}
